// transformer_net_55602646614674
// MI455X (gfx1250) — matrix operand order repaired and checked by reading, not compiled, not run on hardware
//
#include <hip/hip_runtime.h>
#include <math.h>

typedef __attribute__((ext_vector_type(16))) _Float16 v16h;
typedef __attribute__((ext_vector_type(8)))  _Float16 v8h;
typedef __attribute__((ext_vector_type(8)))  float    v8f;
typedef __attribute__((ext_vector_type(4)))  float    v4f;

#define NB 4          // batch
#define NPT 4096      // points (16^3)
#define KNN 7

// ---------------------------------------------------------------------------
// KNN + rel_pos: geometry only, stable top-7 (ties -> lower index, like top_k)
// ---------------------------------------------------------------------------
__global__ void __launch_bounds__(256) knn_kernel(int* __restrict__ knn,
                                                  float* __restrict__ relp) {
  int n = blockIdx.x * blockDim.x + threadIdx.x;
  if (n >= NPT) return;
  int xi = n >> 8, yi = (n >> 4) & 15, zi = n & 15;
  float bd[KNN]; int bi[KNN];
#pragma unroll
  for (int t = 0; t < KNN; ++t) { bd[t] = 3.4e38f; bi[t] = 0; }
  for (int m = 0; m < NPT; ++m) {
    float dx = (float)(xi - (m >> 8));
    float dy = (float)(yi - ((m >> 4) & 15));
    float dz = (float)(zi - (m & 15));
    float dd = dx * dx + dy * dy + dz * dz;
    if (dd < bd[KNN - 1]) {
      int p = KNN - 1;
      while (p > 0 && dd < bd[p - 1]) { bd[p] = bd[p - 1]; bi[p] = bi[p - 1]; --p; }
      bd[p] = dd; bi[p] = m;
    }
  }
#pragma unroll
  for (int t = 0; t < KNN; ++t) {
    int m = bi[t];
    knn[n * KNN + t] = m;
    relp[(n * KNN + t) * 3 + 0] = (float)(xi - (m >> 8));
    relp[(n * KNN + t) * 3 + 1] = (float)(yi - ((m >> 4) & 15));
    relp[(n * KNN + t) * 3 + 2] = (float)(zi - (m & 15));
  }
}

// ---------------------------------------------------------------------------
// pos_enc layer1: (N*K,3) -> (N*K,d), relu   (tiny din=3: VALU)
// ---------------------------------------------------------------------------
__global__ void __launch_bounds__(256) posenc_l1(const float* __restrict__ relp,
                                                 const float* __restrict__ w,
                                                 const float* __restrict__ b,
                                                 float* __restrict__ out, int d) {
  int i = blockIdx.x * blockDim.x + threadIdx.x;
  int total = NPT * KNN * d;
  if (i >= total) return;
  int r = i / d, c = i % d;
  float s = b[c] + relp[r * 3 + 0] * w[c * 3 + 0]
                 + relp[r * 3 + 1] * w[c * 3 + 1]
                 + relp[r * 3 + 2] * w[c * 3 + 2];
  out[i] = fmaxf(s, 0.f);
}

// ---------------------------------------------------------------------------
// fc1 for block 1 (din = 1): x = data * w + b
// ---------------------------------------------------------------------------
__global__ void __launch_bounds__(256) fc1_b1(const float* __restrict__ data,
                                              const float* __restrict__ w,
                                              const float* __restrict__ b,
                                              float* __restrict__ x) {
  int i = blockIdx.x * blockDim.x + threadIdx.x;
  if (i >= NB * NPT * 32) return;
  int r = i >> 5, c = i & 31;
  x[i] = data[r] * w[c] + b[c];
}

// ---------------------------------------------------------------------------
// Generic WMMA GEMM: C[M x dout] = act(A[M x din] @ W[dout x din]^T + bias)
// One wave per 16x16 output tile; f32 global loads -> f16 operands per the
// CDNA5 16-bit A/B VGPR layout; v_wmma_f32_16x16x32_f16 accumulates in f32.
// ---------------------------------------------------------------------------
__global__ void __launch_bounds__(256) wmma_gemm(const float* __restrict__ A,
                                                 const float* __restrict__ W,
                                                 const float* __restrict__ bias,
                                                 float* __restrict__ C,
                                                 int M, int din, int dout, int relu) {
  int wave = threadIdx.x >> 5;
  int tile = blockIdx.x * (blockDim.x >> 5) + wave;
  int mtiles = M >> 4;
  int ntiles = dout >> 4;
  if (tile >= mtiles * ntiles) return;          // wave-uniform exit
  int m0 = (tile % mtiles) << 4;
  int n0 = (tile / mtiles) << 4;
  int lane = threadIdx.x & 31;
  int hi = lane >> 4;
  int l15 = lane & 15;
  const float* arow = A + (size_t)(m0 + l15) * din;
  const float* wrow = W + (size_t)(n0 + l15) * din;
  v8f acc = {};
  for (int kk = 0; kk < din; kk += 32) {
    int ka = kk + (hi ? 8 : 0);
    v4f a0 = *(const v4f*)(arow + ka);
    v4f a1 = *(const v4f*)(arow + ka + 4);
    v4f a2 = *(const v4f*)(arow + ka + 16);
    v4f a3 = *(const v4f*)(arow + ka + 20);
    int kb = kk + (hi ? 8 : 0);
    v4f b0 = *(const v4f*)(wrow + kb);
    v4f b1 = *(const v4f*)(wrow + kb + 4);
    v4f b2 = *(const v4f*)(wrow + kb + 16);
    v4f b3 = *(const v4f*)(wrow + kb + 20);
    v16h av = {}, bv = {};
#pragma unroll
    for (int t = 0; t < 4; ++t) {
      av[t]      = (_Float16)a0[t];
      av[4 + t]  = (_Float16)a1[t];
      av[8 + t]  = (_Float16)a2[t];
      av[12 + t] = (_Float16)a3[t];
      bv[t]      = (_Float16)b0[t];
      bv[4 + t]  = (_Float16)b1[t];
      bv[8 + t]  = (_Float16)b2[t];
      bv[12 + t] = (_Float16)b3[t];
    }
    acc = __builtin_amdgcn_wmma_f32_16x16x32_f16(false, av, false, bv,
                                                 (short)0, acc, false, false);
  }
  float bb = bias ? bias[n0 + l15] : 0.f;
#pragma unroll
  for (int r = 0; r < 8; ++r) {
    float vv = acc[r] + bb;
    if (relu) vv = fmaxf(vv, 0.f);
    C[(size_t)(m0 + r + hi * 8) * dout + (n0 + l15)] = vv;
  }
}

// ---------------------------------------------------------------------------
// LDS-sourced 16x16 WMMA tile (A rows from sA at rowbase, B rows = output
// columns from sB at colbase). Strides are DD+8 halves (16B aligned, bank
// friendly). dk = DD (32 or 64).
// ---------------------------------------------------------------------------
template <int DD>
__device__ inline v8f lds_mm(const _Float16* sA, int rowbase,
                             const _Float16* sB, int colbase) {
  constexpr int ST = DD + 8;
  const int lane = threadIdx.x & 31;
  const int hi = lane >> 4;
  const int l15 = lane & 15;
  v8f acc = {};
#pragma unroll
  for (int kk = 0; kk < DD; kk += 32) {
    const int ka = kk + (hi ? 8 : 0);
    const _Float16* ap = sA + (rowbase + l15) * ST + ka;
    v8h a0 = *(const v8h*)ap;
    v8h a1 = *(const v8h*)(ap + 16);
    const int kb = kk + (hi ? 8 : 0);
    const _Float16* bp = sB + (colbase + l15) * ST + kb;
    v8h b0 = *(const v8h*)bp;
    v8h b1 = *(const v8h*)(bp + 16);
    v16h av = {}, bv = {};
#pragma unroll
    for (int t = 0; t < 8; ++t) {
      av[t] = a0[t]; av[8 + t] = a1[t];
      bv[t] = b0[t]; bv[8 + t] = b1[t];
    }
    acc = __builtin_amdgcn_wmma_f32_16x16x32_f16(false, av, false, bv,
                                                 (short)0, acc, false, false);
  }
  return acc;
}

// ---------------------------------------------------------------------------
// Fused attention block: per workgroup = one batch b and 16 points.
// t(112xDD) -> g1 GEMM+relu -> g2 GEMM (scaled logits) -> softmax over K ->
// weighted sum with (v_gathered + pos_enc) -> fc2 GEMM + residual + BN + ReLU.
// Everything staged in LDS; 8 waves; 7 row-tiles x (DD/16) col-tiles.
// ---------------------------------------------------------------------------
template <int DD>
__global__ void __launch_bounds__(256) attn_kernel(
    const float* __restrict__ xbuf, const float* __restrict__ qbuf,
    const float* __restrict__ kbuf, const float* __restrict__ vbuf,
    const float* __restrict__ pose, const int* __restrict__ knn,
    const float* __restrict__ g1w, const float* __restrict__ g1b,
    const float* __restrict__ g2w, const float* __restrict__ g2b,
    const float* __restrict__ f2w, const float* __restrict__ f2b,
    const float* __restrict__ bng, const float* __restrict__ bnb,
    const float* __restrict__ bnm, const float* __restrict__ bnv,
    float* __restrict__ fout) {
  constexpr int ROWS = 16 * KNN;   // 112
  constexpr int ST = DD + 8;
  constexpr int CT = DD / 16;
  constexpr int TILES = 7 * CT;
  __shared__ __attribute__((aligned(16))) _Float16 sT[ROWS * ST];
  __shared__ __attribute__((aligned(16))) _Float16 sH[ROWS * ST];
  __shared__ __attribute__((aligned(16))) _Float16 sW[DD * ST];
  __shared__ float sBias[DD];
  __shared__ float sRes[16 * DD];
  __shared__ int sIdx[ROWS];

  const int b = blockIdx.x >> 8;
  const int n0 = (blockIdx.x & 255) << 4;
  const int tid = threadIdx.x;
  const int wave = tid >> 5;
  const int lane = tid & 31;
  const int hi = lane >> 4;
  const int l15 = lane & 15;
  const size_t bn0 = (size_t)b * NPT;

  // ---- stage neighbor indices + g1 weights ----
  if (tid < ROWS) sIdx[tid] = knn[(n0 + tid / KNN) * KNN + tid % KNN];
  for (int i = tid; i < DD * DD; i += 256) {
    sW[(i / DD) * ST + (i % DD)] = (_Float16)g1w[i];
  }
  if (tid < DD) sBias[tid] = g1b[tid];
  __syncthreads();

  // ---- build t = q - k_gather + pos_enc (f16 in LDS) ----
  for (int i = tid; i < ROWS * DD; i += 256) {
    int r = i / DD, c = i % DD;
    int nl = r / KNN, j = r % KNN;
    int gi = sIdx[r];
    float t = qbuf[(bn0 + n0 + nl) * DD + c] - kbuf[(bn0 + gi) * DD + c]
            + pose[((size_t)(n0 + nl) * KNN + j) * DD + c];
    sT[r * ST + c] = (_Float16)t;
  }
  __syncthreads();

  // ---- GEMM 1: h = relu(t @ g1^T + b1) -> sH ----
  for (int t = wave; t < TILES; t += 8) {
    int rb = (t % 7) * 16, cb = (t / 7) * 16;
    v8f acc = lds_mm<DD>(sT, rb, sW, cb);
    float bb = sBias[cb + l15];
#pragma unroll
    for (int r = 0; r < 8; ++r) {
      float vv = fmaxf(acc[r] + bb, 0.f);
      sH[(rb + r + hi * 8) * ST + cb + l15] = (_Float16)vv;
    }
  }
  __syncthreads();

  // ---- restage g2 ----
  for (int i = tid; i < DD * DD; i += 256) {
    sW[(i / DD) * ST + (i % DD)] = (_Float16)g2w[i];
  }
  if (tid < DD) sBias[tid] = g2b[tid];
  __syncthreads();

  // ---- GEMM 2: logits = (h @ g2^T + b2) / sqrt(d) -> sT (f16) ----
  const float invs = rsqrtf((float)DD);
  for (int t = wave; t < TILES; t += 8) {
    int rb = (t % 7) * 16, cb = (t / 7) * 16;
    v8f acc = lds_mm<DD>(sH, rb, sW, cb);
    float bb = sBias[cb + l15];
#pragma unroll
    for (int r = 0; r < 8; ++r)
      sT[(rb + r + hi * 8) * ST + cb + l15] = (_Float16)((acc[r] + bb) * invs);
  }
  __syncthreads();

  // ---- softmax over K per (point, channel) + weighted sum ----
  for (int i = tid; i < 16 * DD; i += 256) {
    int nl = i / DD, c = i % DD;
    int rb = nl * KNN;
    float mx = -3.4e38f;
#pragma unroll
    for (int j = 0; j < KNN; ++j) mx = fmaxf(mx, (float)sT[(rb + j) * ST + c]);
    float e[KNN], s = 0.f;
#pragma unroll
    for (int j = 0; j < KNN; ++j) {
      e[j] = expf((float)sT[(rb + j) * ST + c] - mx);
      s += e[j];
    }
    float inv = 1.f / s;
    float r = 0.f;
#pragma unroll
    for (int j = 0; j < KNN; ++j) {
      int gi = sIdx[rb + j];
      float vp = vbuf[(bn0 + gi) * DD + c]
               + pose[((size_t)(n0 + nl) * KNN + j) * DD + c];
      r += e[j] * inv * vp;
    }
    sRes[nl * DD + c] = r;
  }
  __syncthreads();

  // ---- stage res (f16) + fc2 weights ----
  for (int i = tid; i < 16 * DD; i += 256) {
    sH[(i / DD) * ST + (i % DD)] = (_Float16)sRes[i];
  }
  for (int i = tid; i < DD * DD; i += 256) {
    sW[(i / DD) * ST + (i % DD)] = (_Float16)f2w[i];
  }
  if (tid < DD) sBias[tid] = f2b[tid];
  __syncthreads();

  // ---- fc2 GEMM + residual + BN + ReLU -> global ----
  if (wave < CT) {
    int cb = wave * 16;
    v8f acc = lds_mm<DD>(sH, 0, sW, cb);
    int col = cb + l15;
    float bb = sBias[col];
    float mn = bnm[col], iv = rsqrtf(bnv[col] + 1e-5f);
    float gm = bng[col], bt = bnb[col];
#pragma unroll
    for (int r = 0; r < 8; ++r) {
      int nl = r + hi * 8;
      size_t gi = (bn0 + n0 + nl) * DD + col;
      float vv = acc[r] + bb + xbuf[gi];
      vv = (vv - mn) * iv * gm + bt;
      fout[gi] = fmaxf(vv, 0.f);
    }
  }
}

// ---------------------------------------------------------------------------
// Classifier head: central 2x2x2 voxels -> w1+BN+ReLU -> w2. One block per b.
// ---------------------------------------------------------------------------
__global__ void __launch_bounds__(256) cls_kernel(
    const float* __restrict__ f, const float* __restrict__ w1,
    const float* __restrict__ b1, const float* __restrict__ g,
    const float* __restrict__ be, const float* __restrict__ mn,
    const float* __restrict__ vr, const float* __restrict__ w2,
    const float* __restrict__ b2, float* __restrict__ out) {
  int b = blockIdx.x;
  __shared__ float sf[64 * 8];
  __shared__ float sh[64 * 8];
  int tid = threadIdx.x;
  for (int it = tid; it < 512; it += 256) {
    int c = it / 8, t = it % 8;
    int i = 7 + (t >> 2), j = 7 + ((t >> 1) & 1), k = 7 + (t & 1);
    int n = i * 256 + j * 16 + k;
    sf[c * 8 + t] = f[((size_t)b * NPT + n) * 64 + c];
  }
  __syncthreads();
  for (int it = tid; it < 512; it += 256) {
    int o = it / 8, t = it % 8;
    float s = b1[o];
    for (int c = 0; c < 64; ++c) s += w1[o * 64 + c] * sf[c * 8 + t];
    s = (s - mn[o]) * rsqrtf(vr[o] + 1e-5f);
    s = fmaxf(s * g[o] + be[o], 0.f);
    sh[o * 8 + t] = s;
  }
  __syncthreads();
  if (tid < 16) {
    int o2 = tid / 8, t = tid % 8;
    float s = b2[o2];
    for (int o = 0; o < 64; ++o) s += w2[o2 * 64 + o] * sh[o * 8 + t];
    out[b * 16 + o2 * 8 + t] = s;
  }
}

// ---------------------------------------------------------------------------
// Host launcher
// ---------------------------------------------------------------------------
struct TLp {
  const float *fc1_w, *fc1_b, *wq, *wk, *wv, *d1_w, *d1_b, *d2_w, *d2_b,
              *g1_w, *g1_b, *g2_w, *g2_b, *fc2_w, *fc2_b;
};

extern "C" void kernel_launch(void* const* d_in, const int* in_sizes, int n_in,
                              void* d_out, int out_size, void* d_ws, size_t ws_size,
                              hipStream_t stream) {
  (void)in_sizes; (void)n_in; (void)out_size; (void)ws_size;
  auto F = [&](int i) { return (const float*)d_in[i]; };
  auto TL = [&](int b) {
    TLp t = {F(b), F(b + 1), F(b + 2), F(b + 3), F(b + 4), F(b + 5), F(b + 6),
             F(b + 7), F(b + 8), F(b + 9), F(b + 10), F(b + 11), F(b + 12),
             F(b + 13), F(b + 14)};
    return t;
  };
  const float* data = F(0);

  // workspace carve (256B aligned)
  char* wp = (char*)d_ws;
  auto carve = [&](size_t bytes) {
    void* p = wp;
    wp += (bytes + 255) & ~(size_t)255;
    return p;
  };
  int*   knn  = (int*)carve((size_t)NPT * KNN * 4);
  float* relp = (float*)carve((size_t)NPT * KNN * 3 * 4);
  float* tmp  = (float*)carve((size_t)NPT * KNN * 64 * 4);
  float* pose = (float*)carve((size_t)NPT * KNN * 64 * 4);
  float* x    = (float*)carve((size_t)NB * NPT * 64 * 4);
  float* q    = (float*)carve((size_t)NB * NPT * 64 * 4);
  float* kb   = (float*)carve((size_t)NB * NPT * 64 * 4);
  float* vb   = (float*)carve((size_t)NB * NPT * 64 * 4);
  float* fa   = (float*)carve((size_t)NB * NPT * 64 * 4);
  float* fb   = (float*)carve((size_t)NB * NPT * 64 * 4);

  auto gemm = [&](const float* A, const float* W, const float* bias, float* C,
                  int M, int din, int dout, bool relu) {
    int tiles = (M / 16) * (dout / 16);
    int blocks = (tiles + 7) / 8;
    wmma_gemm<<<blocks, 256, 0, stream>>>(A, W, bias, C, M, din, dout,
                                          relu ? 1 : 0);
  };

  auto run_block = [&](TLp p, int din, int d, const float* fin, bool first,
                       const float* bng, const float* bnb, const float* bnm,
                       const float* bnv, float* fo) {
    int t1 = NPT * KNN * d;
    posenc_l1<<<(t1 + 255) / 256, 256, 0, stream>>>(relp, p.d1_w, p.d1_b, tmp, d);
    gemm(tmp, p.d2_w, p.d2_b, pose, NPT * KNN, d, d, false);
    if (first) {
      int t2 = NB * NPT * 32;
      fc1_b1<<<(t2 + 255) / 256, 256, 0, stream>>>(fin, p.fc1_w, p.fc1_b, x);
    } else {
      gemm(fin, p.fc1_w, p.fc1_b, x, NB * NPT, din, d, false);
    }
    gemm(x, p.wq, nullptr, q, NB * NPT, d, d, false);
    gemm(x, p.wk, nullptr, kb, NB * NPT, d, d, false);
    gemm(x, p.wv, nullptr, vb, NB * NPT, d, d, false);
    if (d == 32)
      attn_kernel<32><<<NB * (NPT / 16), 256, 0, stream>>>(
          x, q, kb, vb, pose, knn, p.g1_w, p.g1_b, p.g2_w, p.g2_b, p.fc2_w,
          p.fc2_b, bng, bnb, bnm, bnv, fo);
    else
      attn_kernel<64><<<NB * (NPT / 16), 256, 0, stream>>>(
          x, q, kb, vb, pose, knn, p.g1_w, p.g1_b, p.g2_w, p.g2_b, p.fc2_w,
          p.fc2_b, bng, bnb, bnm, bnv, fo);
  };

  knn_kernel<<<NPT / 256, 256, 0, stream>>>(knn, relp);

  run_block(TL(1),  1,  32, data, true,  F(61), F(62), F(63), F(64), fa);
  run_block(TL(16), 32, 64, fa,   false, F(65), F(66), F(67), F(68), fb);
  run_block(TL(31), 64, 64, fb,   false, F(69), F(70), F(71), F(72), fa);
  run_block(TL(46), 64, 64, fa,   false, F(73), F(74), F(75), F(76), fb);

  cls_kernel<<<NB, 256, 0, stream>>>(fb, F(77), F(78), F(79), F(80), F(81),
                                     F(82), F(83), F(84), (float*)d_out);
}
